// TABlock_26268019982573
// MI455X (gfx1250) — matrix operand order repaired and checked by reading, not compiled, not run on hardware
//
#include <hip/hip_runtime.h>
#include <cmath>

typedef __attribute__((ext_vector_type(16))) _Float16 v16h;
typedef __attribute__((ext_vector_type(8)))  _Float16 v8h;
typedef __attribute__((ext_vector_type(8)))  float    v8f;
typedef __attribute__((ext_vector_type(4)))  unsigned int u32x4;
typedef __attribute__((ext_vector_type(8)))  unsigned int u32x8;

#define DIM   768
#define SEQ   1024
#define BATCH 16
#define NROWS (BATCH * SEQ)   // 16384
#define KSTEPS (DIM / 32)     // 24

// ---- fragment helpers (contiguous k order: element e <-> k = g*16+e, wave32) ----

// A fragment 16x32 f16 from an f32 row-major matrix.
// p = &X[row*ld + kk + g*16]; elements 0..15 <- p[0..15]
static __device__ __forceinline__ v16h cvtA_f32(const float* __restrict__ p) {
    v16h a;
#pragma unroll
    for (int i = 0; i < 8; ++i) a[i] = (_Float16)p[i];
#pragma unroll
    for (int i = 0; i < 8; ++i) a[8 + i] = (_Float16)p[8 + i];
    return a;
}

// B fragment (K x 16): element e -> K = g*16+e, 16 consecutive f32.
static __device__ __forceinline__ v16h cvtB_f32(const float* __restrict__ p) {
    v16h b;
#pragma unroll
    for (int i = 0; i < 16; ++i) b[i] = (_Float16)p[i];
    return b;
}

// A fragment from an f16 row-major matrix (also used for LDS P tile, ld=32).
// p = &X[row*ld + kk + g*16]
static __device__ __forceinline__ v16h loadA_h(const _Float16* __restrict__ p) {
    v16h a;
    v8h lo = *(const v8h*)p;
    v8h hi = *(const v8h*)(p + 8);
#pragma unroll
    for (int i = 0; i < 8; ++i) { a[i] = lo[i]; a[8 + i] = hi[i]; }
    return a;
}

// ---- kernel 1: Y[row,col] = f16( (sum_d x[row,d]*W[col,d] + b[col]) * scale )
// transposed==1: store per-batch transposed layout Y[b][col][row%SEQ] (for V)
__global__ __launch_bounds__(256) void qkv_proj_kernel(
    const float* __restrict__ X, const float* __restrict__ W,
    const float* __restrict__ bias, _Float16* __restrict__ Y,
    float scale, int transposed)
{
    const int lane = threadIdx.x & 31;
    const int wave = threadIdx.x >> 5;
    const int tile = blockIdx.x * 8 + wave;       // 49152 tiles total
    const int mt = tile / (DIM / 16);
    const int nt = tile % (DIM / 16);
    const int g  = lane >> 4;
    const int mn = lane & 15;

    const int arow = mt * 16 + mn;    // A-matrix row held by this lane
    const int wrow = nt * 16 + mn;    // W row (= output feature) for B frag

    v8f c = {};
    for (int kk = 0; kk < DIM; kk += 32) {
        v16h a = cvtA_f32(X + (size_t)arow * DIM + kk + g * 16);
        v16h b = cvtB_f32(W + (size_t)wrow * DIM + kk + g * 16);
        c = __builtin_amdgcn_wmma_f32_16x16x32_f16(false, a, false, b,
                                                   (short)0, c, false, false);
    }

    const int col  = nt * 16 + mn;
    const float bv = bias[col];
    const int row0 = mt * 16 + g * 8;
    if (!transposed) {
#pragma unroll
        for (int r = 0; r < 8; ++r)
            Y[(size_t)(row0 + r) * DIM + col] = (_Float16)((c[r] + bv) * scale);
    } else {
        const int bb = row0 / SEQ;
        const int rk = row0 % SEQ;
        v8h pk;
#pragma unroll
        for (int r = 0; r < 8; ++r) pk[r] = (_Float16)((c[r] + bv) * scale);
        *(v8h*)(Y + (size_t)bb * DIM * SEQ + (size_t)col * SEQ + rk) = pk;
    }
}

// ---- kernel 2: per-row softmax stats (max, sum of exp) via WMMA recompute
__global__ __launch_bounds__(128) void attn_stats_kernel(
    const _Float16* __restrict__ Q,   // [B][SEQ][DIM], pre-scaled by 1/sqrt(D)
    const _Float16* __restrict__ K,   // [B][SEQ][DIM]
    float* __restrict__ Mrow, float* __restrict__ Lrow)
{
    const int lane = threadIdx.x & 31;
    const int wave = threadIdx.x >> 5;
    const int gid  = blockIdx.x * 4 + wave;       // 1024 waves
    const int bb = gid >> 6;
    const int rt = gid & 63;
    const int g  = lane >> 4;
    const int mn = lane & 15;

    const _Float16* Qb = Q + (size_t)bb * SEQ * DIM;
    const _Float16* Kb = K + (size_t)bb * SEQ * DIM;

    // Q fragments resident for the whole key sweep (24 * 8 = 192 VGPRs)
    v16h aq[KSTEPS];
    const int qrow = rt * 16 + mn;
#pragma unroll
    for (int ks = 0; ks < KSTEPS; ++ks)
        aq[ks] = loadA_h(Qb + (size_t)qrow * DIM + ks * 32 + g * 16);

    float m[8], l[8];
#pragma unroll
    for (int r = 0; r < 8; ++r) { m[r] = -1e30f; l[r] = 0.0f; }

    for (int nt = 0; nt < SEQ / 16; ++nt) {
        v8f c = {};
        const int krow = nt * 16 + mn;
#pragma unroll
        for (int ks = 0; ks < KSTEPS; ++ks) {
            v16h bf = *(const v16h*)(Kb + (size_t)krow * DIM + ks * 32 + g * 16);
            c = __builtin_amdgcn_wmma_f32_16x16x32_f16(false, aq[ks], false, bf,
                                                       (short)0, c, false, false);
        }
#pragma unroll
        for (int r = 0; r < 8; ++r) {
            float s  = c[r];
            float nm = fmaxf(m[r], s);
            l[r] = l[r] * __expf(m[r] - nm) + __expf(s - nm);
            m[r] = nm;
        }
    }
    // reduce across the 16 lanes of each half (same rows, different cols)
#pragma unroll
    for (int r = 0; r < 8; ++r) {
        float mm = m[r], ll = l[r];
#pragma unroll
        for (int mask = 1; mask < 16; mask <<= 1) {
            float om = __shfl_xor(mm, mask, 32);
            float ol = __shfl_xor(ll, mask, 32);
            float nm = fmaxf(mm, om);
            ll = ll * __expf(mm - nm) + ol * __expf(om - nm);
            mm = nm;
        }
        m[r] = mm; l[r] = ll;
    }
    if (mn == 0) {
        const int row0 = bb * SEQ + rt * 16 + g * 8;
#pragma unroll
        for (int r = 0; r < 8; ++r) { Mrow[row0 + r] = m[r]; Lrow[row0 + r] = l[r]; }
    }
}

// ---- kernel 3: O = softmax(QK^T) V, fused.
// K tiles (32 keys x 768 dims, 48 KB) are staged into LDS by the Tensor Data
// Mover (one tensor_load_to_lds per chunk, issued by wave 0, TENSORcnt-tracked),
// shared by all 4 waves of the block.
__global__ __launch_bounds__(128) void attn_out_kernel(
    const _Float16* __restrict__ Q,
    const _Float16* __restrict__ K,
    const _Float16* __restrict__ Vt,   // [B][DIM][SEQ] (transposed V)
    const float* __restrict__ Mrow, const float* __restrict__ Lrow,
    const float* __restrict__ X, float* __restrict__ Out)
{
    __shared__ _Float16 ldsK[32 * DIM];       // 48 KB staged K chunk
    __shared__ _Float16 ptile[4][16 * 32];    // 4 KB per-wave P tiles
    const int lane = threadIdx.x & 31;
    const int wave = threadIdx.x >> 5;
    const int gid  = blockIdx.x * 4 + wave;   // 2048 waves: [B][rt][vhalf]
    const int vhalf = gid & 1;
    const int rt    = (gid >> 1) & 63;
    const int bb    = gid >> 7;
    const int g  = lane >> 4;
    const int mn = lane & 15;

    const _Float16* Qb = Q  + (size_t)bb * SEQ * DIM;
    const _Float16* Kb = K  + (size_t)bb * SEQ * DIM;
    const _Float16* Vb = Vt + (size_t)bb * DIM * SEQ;

    const unsigned ldsK_base = (unsigned)(uintptr_t)(&ldsK[0]); // LDS byte offset

    const int row0 = rt * 16 + g * 8;             // query rows of C frags
    float mr[8], li[8];
#pragma unroll
    for (int r = 0; r < 8; ++r) {
        mr[r] = Mrow[bb * SEQ + row0 + r];
        li[r] = 1.0f / Lrow[bb * SEQ + row0 + r];
    }

    const int qrow = rt * 16 + mn;
    v8f zero = {};
    v8f o[24];
#pragma unroll
    for (int t = 0; t < 24; ++t) o[t] = zero;

    _Float16* myP = ptile[wave];

    for (int kc = 0; kc < SEQ / 32; ++kc) {       // 32-key chunks
        const int key0 = kc * 32;
        __syncthreads();                          // ldsK free for reuse
        if (wave == 0) {
            // Build Tensor DMA descriptor (D#), groups 0 and 1 (2D tile).
            unsigned long long ga =
                (unsigned long long)(uintptr_t)(Kb + (size_t)key0 * DIM);
            u32x4 g0;
            g0[0] = 1u;                                   // count=1, user load
            g0[1] = ldsK_base;                            // lds_addr
            g0[2] = (unsigned)ga;                         // global_addr[31:0]
            g0[3] = (unsigned)((ga >> 32) & 0x1FFFFFFu)   // global_addr[56:32]
                    | (2u << 30);                         // type=2 (image)
            u32x8 g1;
            g1[0] = 1u << 16;          // workgroup_mask=0, data_size=1 (2 bytes)
            g1[1] = 768u << 16;        // tensor_dim0[15:0]=768
            g1[2] = 1024u << 16;       // tensor_dim0[31:16]=0, tensor_dim1[15:0]=1024
            g1[3] = 768u << 16;        // tensor_dim1[31:16]=0, tile_dim0=768
            g1[4] = 32u;               // tile_dim1=32, tile_dim2=0
            g1[5] = 768u;              // tensor_dim0_stride[31:0]=768
            g1[6] = 0u;                // stride hi, dim1_stride lo
            g1[7] = 0u;                // dim1_stride hi
            asm volatile("tensor_load_to_lds %0, %1"
                         :: "s"(g0), "s"(g1) : "memory");
            __builtin_amdgcn_s_wait_tensorcnt(0);
        }
        __syncthreads();                          // ldsK ready

        v8f c0 = zero, c1 = zero;
#pragma unroll
        for (int ks = 0; ks < KSTEPS; ++ks) {
            v16h a  = loadA_h(Qb + (size_t)qrow * DIM + ks * 32 + g * 16);
            v16h b0 = *(const v16h*)(ldsK + (size_t)mn        * DIM + ks * 32 + g * 16);
            v16h b1 = *(const v16h*)(ldsK + (size_t)(16 + mn) * DIM + ks * 32 + g * 16);
            c0 = __builtin_amdgcn_wmma_f32_16x16x32_f16(false, a, false, b0,
                                                        (short)0, c0, false, false);
            c1 = __builtin_amdgcn_wmma_f32_16x16x32_f16(false, a, false, b1,
                                                        (short)0, c1, false, false);
        }
        // P = exp(S - m), staged via LDS to relayout C-frag -> A-frag
#pragma unroll
        for (int r = 0; r < 8; ++r) {
            myP[(g * 8 + r) * 32 + mn]      = (_Float16)__expf(c0[r] - mr[r]);
            myP[(g * 8 + r) * 32 + 16 + mn] = (_Float16)__expf(c1[r] - mr[r]);
        }
        asm volatile("s_wait_dscnt 0" ::: "memory");
        v16h p = loadA_h(myP + mn * 32 + g * 16);
        asm volatile("" ::: "memory");
#pragma unroll
        for (int t = 0; t < 24; ++t) {
            const int col = vhalf * 384 + t * 16 + mn;
            v16h bv = *(const v16h*)(Vb + (size_t)col * SEQ + key0 + g * 16);
            o[t] = __builtin_amdgcn_wmma_f32_16x16x32_f16(false, p, false, bv,
                                                          (short)0, o[t], false, false);
        }
        asm volatile("s_wait_dscnt 0" ::: "memory");  // LDS reads drained before rewrite
    }

    // out_flat[b][col*SEQ + row] = av[b,row,col]/l + x_flat[b][col*SEQ + row]
    const float* Xb = X   + (size_t)bb * SEQ * DIM;
    float*       Ob = Out + (size_t)bb * SEQ * DIM;
#pragma unroll
    for (int t = 0; t < 24; ++t) {
        const int col  = vhalf * 384 + t * 16 + mn;
        const size_t p = (size_t)col * SEQ + row0;
        float4 x0 = *(const float4*)(Xb + p);
        float4 x1 = *(const float4*)(Xb + p + 4);
        float4 r0, r1;
        r0.x = o[t][0] * li[0] + x0.x;  r0.y = o[t][1] * li[1] + x0.y;
        r0.z = o[t][2] * li[2] + x0.z;  r0.w = o[t][3] * li[3] + x0.w;
        r1.x = o[t][4] * li[4] + x1.x;  r1.y = o[t][5] * li[5] + x1.y;
        r1.z = o[t][6] * li[6] + x1.z;  r1.w = o[t][7] * li[7] + x1.w;
        *(float4*)(Ob + p)     = r0;
        *(float4*)(Ob + p + 4) = r1;
    }
}

extern "C" void kernel_launch(void* const* d_in, const int* in_sizes, int n_in,
                              void* d_out, int out_size, void* d_ws, size_t ws_size,
                              hipStream_t stream) {
    const float* x  = (const float*)d_in[0];
    const float* Wq = (const float*)d_in[1];
    const float* bq = (const float*)d_in[2];
    const float* Wk = (const float*)d_in[3];
    const float* bk = (const float*)d_in[4];
    const float* Wv = (const float*)d_in[5];
    const float* bv = (const float*)d_in[6];

    // workspace layout (~75.6 MB)
    _Float16* qh = (_Float16*)d_ws;
    _Float16* kh = qh + (size_t)NROWS * DIM;
    _Float16* vh = kh + (size_t)NROWS * DIM;   // stored transposed [B][DIM][SEQ]
    float* Mrow = (float*)(vh + (size_t)NROWS * DIM);
    float* Lrow = Mrow + NROWS;

    const float scale = 1.0f / sqrtf((float)DIM);

    // 49152 tiles / 8 waves per block
    qkv_proj_kernel<<<6144, 256, 0, stream>>>(x, Wq, bq, qh, scale, 0);
    qkv_proj_kernel<<<6144, 256, 0, stream>>>(x, Wk, bk, kh, 1.0f, 0);
    qkv_proj_kernel<<<6144, 256, 0, stream>>>(x, Wv, bv, vh, 1.0f, 1);

    attn_stats_kernel<<<256, 128, 0, stream>>>(qh, kh, Mrow, Lrow);
    attn_out_kernel<<<512, 128, 0, stream>>>(qh, kh, vh, Mrow, Lrow, x, (float*)d_out);
}
